// MultiheadCrossAttention_8443905704416
// MI455X (gfx1250) — matrix operand order repaired and checked by reading, not compiled, not run on hardware
//
#include <hip/hip_runtime.h>

#define DMODEL 512
#define HEADS  8
#define HD     64
#define BB     4
#define TT     2048
#define SS     4096

#define KT_STRIDE 72   // bf16 elems per K-tile row  (144B, 36 dwords: conflict-free)
#define VT_STRIDE 40   // bf16 elems per V-tile row  ( 80B, 20 dwords: conflict-free)
#define PT_STRIDE 40   // bf16 elems per P-tile row

typedef __attribute__((ext_vector_type(16))) __bf16 v16bf;
typedef __attribute__((ext_vector_type(8)))  __bf16 v8bf;
typedef __attribute__((ext_vector_type(8)))  float  v8f;

// xor-butterfly via ds_swizzle (group-of-32 mode: and=0x1f, or=0, xor=M).
template <int M>
static __device__ __forceinline__ float swz_xor(float x) {
  return __int_as_float(
      __builtin_amdgcn_ds_swizzle(__float_as_int(x), (M << 10) | 0x1f));
}
static __device__ __forceinline__ float red_max16(float v) {
  v = fmaxf(v, swz_xor<1>(v));
  v = fmaxf(v, swz_xor<2>(v));
  v = fmaxf(v, swz_xor<4>(v));
  v = fmaxf(v, swz_xor<8>(v));
  return v;
}
static __device__ __forceinline__ float red_sum16(float v) {
  v += swz_xor<1>(v);
  v += swz_xor<2>(v);
  v += swz_xor<4>(v);
  v += swz_xor<8>(v);
  return v;
}

// Async DMA: 16 bytes global -> LDS per lane (ASYNCcnt-tracked).
// LDS address operand = low 32 bits of the generic shared pointer (aperture rule).
static __device__ __forceinline__ void async_ld16(void* lds, const void* gsrc) {
  unsigned l = (unsigned)(uintptr_t)lds;
  asm volatile("global_load_async_to_lds_b128 %0, %1, off"
               :: "v"(l), "v"(gsrc) : "memory");
}

// 16x32 bf16 A-fragment: lane -> row = L%16, half = L/16;
// elem j: K = 16*half+j.
static __device__ __forceinline__ v16bf mk_a_frag(const __bf16* rowptr, int half) {
  const v8bf lo = *(const v8bf*)(rowptr + 16 * half);
  const v8bf hi = *(const v8bf*)(rowptr + 16 * half + 8);
  v16bf a;
#pragma unroll
  for (int j = 0; j < 8; ++j) { a[j] = lo[j]; a[8 + j] = hi[j]; }
  return a;
}

// Generic C = A(f32,MxK) * W(f32,KxN) + bias, bf16 WMMA, f32 accumulate.
// MODE 0: out0 = Q bf16 (B,H,rows,64), pre-scaled by 1/sqrt(HD)
// MODE 1: out0 = K bf16 (B,H,rows,64); out1 = V^T bf16 (B,H,64,rows)
// MODE 2: outf = plain f32 row-major MxN
template <int MODE>
__global__ __launch_bounds__(128, 1)
void gemm_bias_wmma(const float* __restrict__ A, const float* __restrict__ W,
                    const float* __restrict__ bias,
                    int M, int N, int K, int rowsPerBatch,
                    __bf16* __restrict__ out0, __bf16* __restrict__ out1,
                    float* __restrict__ outf) {
  __shared__ __align__(64) __bf16 Wl[32 * 64];
  const int tid  = threadIdx.x;
  const int lane = tid & 31;
  const int wave = tid >> 5;
  const int half = lane >> 4;
  const int lm   = lane & 15;
  const int m0 = (blockIdx.x * 4 + wave) * 16;
  const int n0 = blockIdx.y * 64;

  v8f acc[4] = {};
  const float* arow = A + (size_t)(m0 + lm) * K;

  for (int k0 = 0; k0 < K; k0 += 32) {
    {
      const int row = tid >> 2;
      const int c0  = (tid & 3) * 16;
      const float* wsrc = W + (size_t)(k0 + row) * N + n0 + c0;
      __bf16* wdst = Wl + row * 64 + c0;
#pragma unroll
      for (int j = 0; j < 16; ++j) wdst[j] = (__bf16)wsrc[j];
    }
    __syncthreads();

    v16bf a;
#pragma unroll
    for (int j = 0; j < 8; ++j) {
      a[j]     = (__bf16)arow[k0 + 16 * half + j];
      a[8 + j] = (__bf16)arow[k0 + 16 * half + 8 + j];
    }
#pragma unroll
    for (int nt = 0; nt < 4; ++nt) {
      v16bf bw;
#pragma unroll
      for (int j = 0; j < 16; ++j)
        bw[j] = Wl[(16 * half + j) * 64 + nt * 16 + lm];
      acc[nt] = __builtin_amdgcn_wmma_f32_16x16x32_bf16(
          false, a, false, bw, (short)0, acc[nt], false, false);
    }
    __syncthreads();
  }

#pragma unroll
  for (int nt = 0; nt < 4; ++nt) {
    const int n = n0 + nt * 16 + lm;
    const float bv = bias[n];
#pragma unroll
    for (int r = 0; r < 8; ++r) {
      const int m = m0 + r + 8 * half;
      const float val = acc[nt][r] + bv;
      if (MODE == 0) {
        const int b = m / rowsPerBatch, t = m % rowsPerBatch;
        const int h = n >> 6, d = n & 63;
        out0[(((size_t)b * HEADS + h) * rowsPerBatch + t) * HD + d] =
            (__bf16)(val * 0.125f);
      } else if (MODE == 1) {
        const int b = m / rowsPerBatch, s = m % rowsPerBatch;
        if (n < DMODEL) {
          const int h = n >> 6, d = n & 63;
          out0[(((size_t)b * HEADS + h) * rowsPerBatch + s) * HD + d] = (__bf16)val;
        } else {
          const int h = (n - DMODEL) >> 6, d = n & 63;
          out1[(((size_t)b * HEADS + h) * HD + d) * (size_t)rowsPerBatch + s] = (__bf16)val;
        }
      } else {
        outf[(size_t)m * N + n] = val;
      }
    }
  }
}

// Flash attention: 8 waves (256 threads) per block, all on one (b,h).
// Each wave owns a 16-row T tile; K/V^T tiles are async-DMA'd into LDS,
// double-buffered, and shared by all 8 waves (8x L2 traffic reduction).
__global__ __launch_bounds__(256, 1)
void attn_flash_wmma(const __bf16* __restrict__ Qb, const __bf16* __restrict__ Kb,
                     const __bf16* __restrict__ Vt, const int* __restrict__ mask,
                     float* __restrict__ ctx) {
  __shared__ __align__(64) __bf16 Kt[2][32 * KT_STRIDE];   // 9216 B
  __shared__ __align__(64) __bf16 Vl[2][64 * VT_STRIDE];   // 10240 B
  __shared__ __align__(64) __bf16 Pl[8][16 * PT_STRIDE];   // 10240 B

  const int tid  = threadIdx.x;
  const int lane = tid & 31;
  const int wave = tid >> 5;
  const int half = lane >> 4;
  const int lm   = lane & 15;

  const int tBlocks = TT / 128;                 // 16 blocks per (b,h)
  const int bh   = blockIdx.x / tBlocks;
  const int tblk = blockIdx.x % tBlocks;
  const int t0   = tblk * 128 + wave * 16;
  const int b = bh / HEADS, h = bh % HEADS;

  const __bf16* qrow = Qb + ((size_t)bh * TT + (t0 + lm)) * HD;
  const v16bf aq0 = mk_a_frag(qrow, half);
  const v16bf aq1 = mk_a_frag(qrow + 32, half);

  float mstat[8], lstat[8];
#pragma unroll
  for (int r = 0; r < 8; ++r) { mstat[r] = -1e30f; lstat[r] = 0.f; }
  v8f oacc[4] = {};

  __bf16* pl = Pl[wave];
  const int* mrow = mask + b * SS;

  // stage K[ s0 : s0+32, 0:64 ] and V^T[ 0:64, s0 : s0+32 ] into buffer `buf`
  auto stage = [&](int buf, int s0) {
    {   // K: 32 rows x 64 el, contiguous per row; 256 thr x 8 el
      const int row = tid >> 3, c = (tid & 7) * 8;
      async_ld16(&Kt[buf][row * KT_STRIDE + c],
                 Kb + ((size_t)bh * SS + s0 + row) * HD + c);
    }
    {   // V^T: 64 rows x 32 el; 256 thr x 8 el
      const int row = tid >> 2, c = (tid & 3) * 8;
      async_ld16(&Vl[buf][row * VT_STRIDE + c],
                 Vt + ((size_t)bh * HD + row) * SS + s0 + c);
    }
  };

  stage(0, 0);

  const int nIter = SS / 32;                    // 128
  for (int i = 0; i < nIter; ++i) {
    const int s0  = i * 32;
    const int buf = i & 1;
    if (i + 1 < nIter) {
      stage(buf ^ 1, s0 + 32);
      asm volatile("s_wait_asynccnt 0x2" ::: "memory");  // stage(i) done
    } else {
      asm volatile("s_wait_asynccnt 0x0" ::: "memory");
    }
    __syncthreads();                            // all waves' stage(i) visible

    const __bf16* kt = Kt[buf];
    const __bf16* vl = Vl[buf];

    v8f scA = {}, scB = {};
    {   // B-frag (K^T 64x16): lane -> s = lm; elem j -> d = kk + 16*half + j
      const v16bf bk0 = *(const v16bf*)(kt + lm * KT_STRIDE + 16 * half);
      const v16bf bk1 = *(const v16bf*)(kt + lm * KT_STRIDE + 32 + 16 * half);
      scA = __builtin_amdgcn_wmma_f32_16x16x32_bf16(false, aq0, false, bk0, (short)0, scA, false, false);
      scA = __builtin_amdgcn_wmma_f32_16x16x32_bf16(false, aq1, false, bk1, (short)0, scA, false, false);
    }
    {
      const v16bf bk0 = *(const v16bf*)(kt + (16 + lm) * KT_STRIDE + 16 * half);
      const v16bf bk1 = *(const v16bf*)(kt + (16 + lm) * KT_STRIDE + 32 + 16 * half);
      scB = __builtin_amdgcn_wmma_f32_16x16x32_bf16(false, aq0, false, bk0, (short)0, scB, false, false);
      scB = __builtin_amdgcn_wmma_f32_16x16x32_bf16(false, aq1, false, bk1, (short)0, scB, false, false);
    }
    const bool okA = mrow[s0 + lm] != 0;
    const bool okB = mrow[s0 + 16 + lm] != 0;

    // Online softmax, per-row to minimize live registers.
#pragma unroll
    for (int r = 0; r < 8; ++r) {               // row = t0 + r + 8*half
      const float a   = okA ? scA[r] : -1e30f;  // Q already carries 1/sqrt(hd)
      const float bsc = okB ? scB[r] : -1e30f;
      const float mnew  = fmaxf(mstat[r], red_max16(fmaxf(a, bsc)));
      const float alpha = __expf(mstat[r] - mnew);
      mstat[r] = mnew;
      const float pA = (a   < -1e29f) ? 0.f : __expf(a   - mnew);
      const float pB = (bsc < -1e29f) ? 0.f : __expf(bsc - mnew);
      lstat[r] = lstat[r] * alpha + red_sum16(pA + pB);
      oacc[0][r] *= alpha;
      oacc[1][r] *= alpha;
      oacc[2][r] *= alpha;
      oacc[3][r] *= alpha;
      pl[(r + 8 * half) * PT_STRIDE + lm]      = (__bf16)pA;
      pl[(r + 8 * half) * PT_STRIDE + 16 + lm] = (__bf16)pB;
    }
    asm volatile("s_wait_dscnt 0x0" ::: "memory");
    const v16bf ap = mk_a_frag(pl + lm * PT_STRIDE, half);

#pragma unroll
    for (int nt = 0; nt < 4; ++nt) {
      // B-frag from V^T tile: lane -> d = nt*16+lm; elem j -> s = 16*half + j
      const v16bf bv = *(const v16bf*)(vl + (nt * 16 + lm) * VT_STRIDE + 16 * half);
      oacc[nt] = __builtin_amdgcn_wmma_f32_16x16x32_bf16(false, ap, false, bv, (short)0, oacc[nt], false, false);
    }
    __syncthreads();                            // done reading buf before restage
  }

#pragma unroll
  for (int r = 0; r < 8; ++r) {
    const float inv = lstat[r] > 0.f ? 1.f / lstat[r] : 0.f;
    const int t = t0 + r + 8 * half;
#pragma unroll
    for (int nt = 0; nt < 4; ++nt)
      ctx[((size_t)b * TT + t) * DMODEL + h * HD + nt * 16 + lm] = oacc[nt][r] * inv;
  }
}

extern "C" void kernel_launch(void* const* d_in, const int* in_sizes, int n_in,
                              void* d_out, int out_size, void* d_ws, size_t ws_size,
                              hipStream_t stream) {
  const float* x      = (const float*)d_in[0];
  const float* memory = (const float*)d_in[1];
  const int*   mmask  = (const int*)d_in[2];
  const float* Wq  = (const float*)d_in[3];
  const float* bq  = (const float*)d_in[4];
  const float* Wkv = (const float*)d_in[5];
  const float* bkv = (const float*)d_in[6];
  const float* Wo  = (const float*)d_in[7];
  const float* bo  = (const float*)d_in[8];
  float* out = (float*)d_out;

  char* p = (char*)d_ws;
  __bf16* Qb = (__bf16*)p; p += (size_t)BB * HEADS * TT * HD * 2;   // 8 MB
  __bf16* Kb = (__bf16*)p; p += (size_t)BB * HEADS * SS * HD * 2;   // 16 MB
  __bf16* Vt = (__bf16*)p; p += (size_t)BB * HEADS * SS * HD * 2;   // 16 MB
  float* ctx = (float*)p;                                           // 16 MB

  const int MQ  = BB * TT;   // 8192
  const int MKV = BB * SS;   // 16384

  gemm_bias_wmma<0><<<dim3(MQ / 64, DMODEL / 64), dim3(128), 0, stream>>>(
      x, Wq, bq, MQ, DMODEL, DMODEL, TT, Qb, nullptr, nullptr);
  gemm_bias_wmma<1><<<dim3(MKV / 64, (2 * DMODEL) / 64), dim3(128), 0, stream>>>(
      memory, Wkv, bkv, MKV, 2 * DMODEL, DMODEL, SS, Kb, Vt, nullptr);
  attn_flash_wmma<<<dim3(BB * HEADS * (TT / 128)), dim3(256), 0, stream>>>(
      Qb, Kb, Vt, mmask, ctx);
  gemm_bias_wmma<2><<<dim3(MQ / 64, DMODEL / 64), dim3(128), 0, stream>>>(
      ctx, Wo, bo, MQ, DMODEL, DMODEL, TT, nullptr, nullptr, out);
}
